// CoRoPEAttention_57698590654626
// MI455X (gfx1250) — compile-verified, matrix operand order checked by reading, not run on hardware
//
#include <hip/hip_runtime.h>
#include <hip/hip_bf16.h>

typedef __attribute__((ext_vector_type(16))) _Float16 v16h;
typedef __attribute__((ext_vector_type(8)))  float    v8f;

#define B_   2
#define S_   2048
#define E_   1024
#define H_   16
#define D_   64
#define BS_  (B_ * S_)                       // 4096
#define BHSD_ ((size_t)B_ * H_ * S_ * D_)    // 4194304

// K-position of element pair (vi, half) inside a 16-bit A/B WMMA fragment:
// lanes 0-15 hold K 0-7 (v0..v3) and 16-23 (v4..v7); lanes 16-31 hold +8.
__device__ __forceinline__ int kpos(int vi, int kh) {
  return ((vi < 4) ? (2 * vi) : (16 + 2 * (vi - 4))) + kh;
}

__device__ __forceinline__ v8f wmma16(v16h a, v16h b, v8f c) {
  return __builtin_amdgcn_wmma_f32_16x16x32_f16(false, a, false, b, (short)0, c,
                                                false, false);
}

// CDNA5 async copy: global -> LDS, 16 bytes per lane, tracked by ASYNCcnt.
__device__ __forceinline__ void async_load_b128(const void* gptr, void* lptr) {
  asm volatile("global_load_async_to_lds_b128 %0, %1, off"
               :: "v"((unsigned)(uintptr_t)lptr),
                  "v"((unsigned long long)(uintptr_t)gptr)
               : "memory");
}
__device__ __forceinline__ void wait_async0() {
  asm volatile("s_wait_asynccnt 0x0" ::: "memory");
}

// ---------------------------------------------------------------------------
// fp32 -> f16 bulk conversion (vectorized, n % 1024 == 0)
// ---------------------------------------------------------------------------
__global__ __launch_bounds__(256)
void f32_to_f16_kernel(const float* __restrict__ in, _Float16* __restrict__ out) {
  const int i = (blockIdx.x * 256 + threadIdx.x) * 4;
  const float4 f = *reinterpret_cast<const float4*>(in + i);
  union { _Float16 h[4]; uint2 u; } cv;
  cv.h[0] = (_Float16)f.x; cv.h[1] = (_Float16)f.y;
  cv.h[2] = (_Float16)f.z; cv.h[3] = (_Float16)f.w;
  *reinterpret_cast<uint2*>(out + i) = cv.u;
}

// ---------------------------------------------------------------------------
// Y = X(M,K) * W(N,K)^T + bias.  X,W are f16; tiles staged to LDS with
// global_load_async_to_lds_b128.
// MODE 0: fp32, row-major (M,N)          (final projection -> d_out)
// MODE 1: fp32, scattered to (B,H,S,D)   (Q, K kept fp32 for gate/rope)
// MODE 2: f16,  scattered to (B,H,S,D)   (V, straight into WMMA format)
// Block: 256 thr (8 waves), tile 128x128, K-step 32, wave = 32x64 (2x4 tiles).
// ---------------------------------------------------------------------------
template <int MODE>
__global__ __launch_bounds__(256)
void gemm_xwT_kernel(const _Float16* __restrict__ X, const _Float16* __restrict__ W,
                     const float* __restrict__ bias, void* __restrict__ Y,
                     int M, int N, int K) {
  __shared__ _Float16 As[128][40];   // row stride 80 B (16B-aligned)
  __shared__ _Float16 Bs[128][40];

  const int n0   = blockIdx.x * 128;
  const int m0   = blockIdx.y * 128;
  const int tid  = threadIdx.x;
  const int lane = tid & 31;
  const int wid  = tid >> 5;
  const int wm   = (wid & 3) * 32;   // wave row origin in tile
  const int wn   = (wid >> 2) * 64;  // wave col origin in tile
  const int lm   = lane & 15;
  const int kh   = (lane >> 4) * 8;

  const v8f zero = {0.f, 0.f, 0.f, 0.f, 0.f, 0.f, 0.f, 0.f};
  v8f acc[2][4];
#pragma unroll
  for (int i = 0; i < 2; ++i)
#pragma unroll
    for (int j = 0; j < 4; ++j) acc[i][j] = zero;

  for (int k0 = 0; k0 < K; k0 += 32) {
    // 128x32 halves = 512 chunks of 16B per tile; 2 chunks/thread/tile.
#pragma unroll
    for (int rep = 0; rep < 2; ++rep) {
      const int cid  = tid + rep * 256;
      const int row  = cid >> 2;
      const int col8 = (cid & 3) * 8;
      async_load_b128(X + (size_t)(m0 + row) * K + k0 + col8, &As[row][col8]);
      async_load_b128(W + (size_t)(n0 + row) * K + k0 + col8, &Bs[row][col8]);
    }
    wait_async0();
    __syncthreads();

    v16h a[2], bfr[4];
#pragma unroll
    for (int tm = 0; tm < 2; ++tm) {
      const int row = wm + tm * 16 + lm;
#pragma unroll
      for (int vi = 0; vi < 8; ++vi) {
        const int k = kpos(vi, kh);
        a[tm][2 * vi]     = As[row][k];
        a[tm][2 * vi + 1] = As[row][k + 1];
      }
    }
#pragma unroll
    for (int tn = 0; tn < 4; ++tn) {
      const int row = wn + tn * 16 + lm;
#pragma unroll
      for (int vi = 0; vi < 8; ++vi) {
        const int k = kpos(vi, kh);
        bfr[tn][2 * vi]     = Bs[row][k];
        bfr[tn][2 * vi + 1] = Bs[row][k + 1];
      }
    }
#pragma unroll
    for (int tm = 0; tm < 2; ++tm)
#pragma unroll
      for (int tn = 0; tn < 4; ++tn)
        acc[tm][tn] = wmma16(a[tm], bfr[tn], acc[tm][tn]);
    __syncthreads();   // protect As/Bs WAR before next iteration's asyncs
  }

  const int nl = lane & 15;
  const int mb = (lane >> 4) * 8;
#pragma unroll
  for (int tm = 0; tm < 2; ++tm)
#pragma unroll
    for (int tn = 0; tn < 4; ++tn)
#pragma unroll
      for (int vi = 0; vi < 8; ++vi) {
        const int gm = m0 + wm + tm * 16 + mb + vi;
        const int gn = n0 + wn + tn * 16 + nl;
        const float val = acc[tm][tn][vi] + bias[gn];
        if (MODE == 0) {
          ((float*)Y)[(size_t)gm * N + gn] = val;
        } else {
          const int bb = gm / S_;
          const int s  = gm - bb * S_;
          const int h  = gn >> 6;
          const int d  = gn & 63;
          const size_t idx = (((size_t)bb * H_ + h) * S_ + s) * D_ + d;
          if (MODE == 1) ((float*)Y)[idx] = val;
          else           ((_Float16*)Y)[idx] = (_Float16)val;
        }
      }
}

// ---------------------------------------------------------------------------
// gates[b,h,s] = sigmoid(q.k / 8); a_k = cumsum_s(gates).  One block per (b,h).
// ---------------------------------------------------------------------------
__global__ __launch_bounds__(256)
void gates_cumsum_kernel(const float* __restrict__ q, const float* __restrict__ k,
                         float* __restrict__ a_k) {
  __shared__ float part[256];
  const int bh = blockIdx.x;
  const int t  = threadIdx.x;
  const size_t base = (size_t)bh * S_ * D_;

  float vals[8];
  float run = 0.f;
#pragma unroll
  for (int i = 0; i < 8; ++i) {
    const int s = t * 8 + i;
    const float4* qp = (const float4*)(q + base + (size_t)s * D_);
    const float4* kp = (const float4*)(k + base + (size_t)s * D_);
    float acc = 0.f;
#pragma unroll
    for (int j = 0; j < 16; ++j) {
      const float4 a = qp[j], b = kp[j];
      acc += a.x * b.x + a.y * b.y + a.z * b.z + a.w * b.w;
    }
    run += 1.f / (1.f + __expf(-acc * 0.125f));
    vals[i] = run;
  }
  part[t] = run;
  __syncthreads();
  if (t == 0) {
    float p = 0.f;
    for (int i = 0; i < 256; ++i) { const float c = part[i]; part[i] = p; p += c; }
  }
  __syncthreads();
  const float off = part[t];
#pragma unroll
  for (int i = 0; i < 8; ++i)
    a_k[(size_t)bh * S_ + t * 8 + i] = off + vals[i];
}

// ---------------------------------------------------------------------------
// Contextual RoPE: one thread per pair (b,h,s,i), i in [0,32).
// ---------------------------------------------------------------------------
__global__ __launch_bounds__(256)
void rope_kernel(const float* __restrict__ q, const float* __restrict__ k,
                 const float* __restrict__ a_k,
                 _Float16* __restrict__ qr, _Float16* __restrict__ kr) {
  const int gid  = blockIdx.x * 256 + threadIdx.x;
  const int i    = gid & 31;
  const int sidx = gid >> 5;  // (b*H+h)*S + s
  // freq = THETA^(-i/32) = exp(-i * ln(10000)/32)
  const float ang = a_k[sidx] * __expf((float)i * (-9.210340371976184f / 32.f));
  const float c = __cosf(ang), sn = __sinf(ang);
  const size_t base = (size_t)sidx * D_ + 2 * i;
  const float q0 = q[base], q1 = q[base + 1];
  qr[base]     = (_Float16)(q0 * c - q1 * sn);
  qr[base + 1] = (_Float16)(q1 * c + q0 * sn);
  const float k0 = k[base], k1 = k[base + 1];
  kr[base]     = (_Float16)(k0 * c - k1 * sn);
  kr[base + 1] = (_Float16)(k1 * c + k0 * sn);
}

// ---------------------------------------------------------------------------
// Causal flash attention.  Block = 8 waves, 16 query rows/wave (128 rows/blk).
// Uniform key loop across the block; K/V tiles (32x64 f16 = 4KB each) staged
// cooperatively via global_load_async_to_lds_b128 (1 b128 chunk per thread
// per tile), then all 8 waves read fragments from LDS.
// P C-layout -> A-layout goes through a wave-private LDS patch (LDS ops from
// one wave are processed in order, so no barrier needed for that transpose).
// Output written as f16 (B,S,E) for the final projection GEMM.
// ---------------------------------------------------------------------------
__global__ __launch_bounds__(256)
void flash_attn_kernel(const _Float16* __restrict__ qr,
                       const _Float16* __restrict__ kr,
                       const _Float16* __restrict__ v,
                       _Float16* __restrict__ o) {
  __shared__ _Float16 Ks[32][72];    // row stride 144 B (16B-aligned)
  __shared__ _Float16 Vs[32][72];
  __shared__ _Float16 Plds[8][16][34];

  const int bh   = blockIdx.y;
  const int bb   = bh >> 4;  // H_ = 16
  const int h    = bh & 15;
  const int tid  = threadIdx.x;
  const int lane = tid & 31;
  const int wid  = tid >> 5;
  const int lm   = lane & 15;
  const int kh   = (lane >> 4) * 8;
  const int nl   = lane & 15;
  const int mb   = (lane >> 4) * 8;
  const int m0   = blockIdx.x * 128 + wid * 16;
  const size_t base = (size_t)bh * S_ * D_;

  // staging chunk for this thread: one 16B chunk of each 32x64 tile
  const int skey = tid >> 3;         // 0..31
  const int sc8  = (tid & 7) * 8;    // half-offset 0..56

  const v8f zero = {0.f, 0.f, 0.f, 0.f, 0.f, 0.f, 0.f, 0.f};

  // Resident Q fragments: 16 rows x 64 (two K=32 chunks)
  v16h aq[2];
#pragma unroll
  for (int c = 0; c < 2; ++c) {
    const _Float16* qp = qr + base + (size_t)(m0 + lm) * D_ + c * 32;
#pragma unroll
    for (int vi = 0; vi < 8; ++vi) {
      const int kk = kpos(vi, kh);
      aq[c][2 * vi]     = qp[kk];
      aq[c][2 * vi + 1] = qp[kk + 1];
    }
  }

  v8f oa[4] = {zero, zero, zero, zero};
  float row_m[8], row_l[8];
#pragma unroll
  for (int i = 0; i < 8; ++i) { row_m[i] = -1e30f; row_l[i] = 0.f; }

  const int kend = (blockIdx.x + 1) * 128;
  for (int kb = 0; kb < kend; kb += 32) {
    // ---- async stage K and V tiles into LDS ----
    async_load_b128(kr + base + (size_t)(kb + skey) * D_ + sc8, &Ks[skey][sc8]);
    async_load_b128(v  + base + (size_t)(kb + skey) * D_ + sc8, &Vs[skey][sc8]);
    wait_async0();
    __syncthreads();

    // ---- scores = Q K^T ----
    v8f s0 = zero, s1 = zero;
#pragma unroll
    for (int nt = 0; nt < 2; ++nt) {
#pragma unroll
      for (int c = 0; c < 2; ++c) {
        v16h bk;
#pragma unroll
        for (int vi = 0; vi < 8; ++vi) {
          const int kk = kpos(vi, kh);
          bk[2 * vi]     = Ks[nt * 16 + nl][c * 32 + kk];
          bk[2 * vi + 1] = Ks[nt * 16 + nl][c * 32 + kk + 1];
        }
        if (nt == 0) s0 = wmma16(aq[c], bk, s0);
        else         s1 = wmma16(aq[c], bk, s1);
      }
    }
    // scale + causal mask (element (row=mb+vi, col=kb{+16}+nl))
#pragma unroll
    for (int vi = 0; vi < 8; ++vi) {
      const int row = m0 + mb + vi;
      float x0 = s0[vi] * 0.125f;
      float x1 = s1[vi] * 0.125f;
      if (kb + nl > row)      x0 = -1e30f;
      if (kb + 16 + nl > row) x1 = -1e30f;
      s0[vi] = x0; s1[vi] = x1;
    }
    // online softmax (row group = 16 lanes; xor 1/2/4/8 stays in group)
#pragma unroll
    for (int vi = 0; vi < 8; ++vi) {
      float mx = fmaxf(s0[vi], s1[vi]);
      mx = fmaxf(mx, __shfl_xor(mx, 1));
      mx = fmaxf(mx, __shfl_xor(mx, 2));
      mx = fmaxf(mx, __shfl_xor(mx, 4));
      mx = fmaxf(mx, __shfl_xor(mx, 8));
      const float mn    = fmaxf(row_m[vi], mx);
      const float alpha = __expf(row_m[vi] - mn);
      row_m[vi] = mn;
      const float p0 = __expf(s0[vi] - mn);
      const float p1 = __expf(s1[vi] - mn);
      s0[vi] = p0; s1[vi] = p1;
      float rs = p0 + p1;
      rs += __shfl_xor(rs, 1);
      rs += __shfl_xor(rs, 2);
      rs += __shfl_xor(rs, 4);
      rs += __shfl_xor(rs, 8);
      row_l[vi] = row_l[vi] * alpha + rs;
#pragma unroll
      for (int t2 = 0; t2 < 4; ++t2) oa[t2][vi] *= alpha;
    }
    // P: C-layout -> wave-private LDS patch -> A-layout fragment
#pragma unroll
    for (int vi = 0; vi < 8; ++vi) {
      Plds[wid][mb + vi][nl]      = (_Float16)s0[vi];
      Plds[wid][mb + vi][16 + nl] = (_Float16)s1[vi];
    }
    v16h pa;
#pragma unroll
    for (int vi = 0; vi < 8; ++vi) {
      const int kk = kpos(vi, kh);
      pa[2 * vi]     = Plds[wid][lm][kk];
      pa[2 * vi + 1] = Plds[wid][lm][kk + 1];
    }
    // ---- O += P * V  (4 d-tiles of 16) ----
#pragma unroll
    for (int t2 = 0; t2 < 4; ++t2) {
      v16h bv;
#pragma unroll
      for (int vi = 0; vi < 8; ++vi) {
        const int kk = kpos(vi, kh);
        bv[2 * vi]     = Vs[kk][t2 * 16 + nl];
        bv[2 * vi + 1] = Vs[kk + 1][t2 * 16 + nl];
      }
      oa[t2] = wmma16(pa, bv, oa[t2]);
    }
    __syncthreads();   // all waves done with Ks/Vs before next asyncs land
  }

  // normalize + write f16 into (B,S,E)
#pragma unroll
  for (int t2 = 0; t2 < 4; ++t2)
#pragma unroll
    for (int vi = 0; vi < 8; ++vi) {
      const int gm = m0 + mb + vi;
      const int d  = t2 * 16 + nl;
      const float val = oa[t2][vi] / row_l[vi];
      o[((size_t)bb * S_ + gm) * E_ + h * D_ + d] = (_Float16)val;
    }
}

// ---------------------------------------------------------------------------
extern "C" void kernel_launch(void* const* d_in, const int* in_sizes, int n_in,
                              void* d_out, int out_size, void* d_ws, size_t ws_size,
                              hipStream_t stream) {
  const float* x  = (const float*)d_in[0];
  const float* Wq = (const float*)d_in[1];
  const float* bq = (const float*)d_in[2];
  const float* Wk = (const float*)d_in[3];
  const float* bk = (const float*)d_in[4];
  const float* Wv = (const float*)d_in[5];
  const float* bv = (const float*)d_in[6];
  const float* Wo = (const float*)d_in[7];
  const float* bo = (const float*)d_in[8];

  char* ws = (char*)d_ws;
  float*    q32  = (float*)ws;     ws += BHSD_ * sizeof(float);            // 16 MB
  float*    k32  = (float*)ws;     ws += BHSD_ * sizeof(float);            // 16 MB
  _Float16* v16p = (_Float16*)ws;  ws += BHSD_ * sizeof(_Float16);         //  8 MB
  float*    ak   = (float*)ws;     ws += (size_t)B_ * H_ * S_ * sizeof(float);
  _Float16* qrp  = (_Float16*)ws;  ws += BHSD_ * sizeof(_Float16);         //  8 MB
  _Float16* krp  = (_Float16*)ws;  ws += BHSD_ * sizeof(_Float16);         //  8 MB
  _Float16* obuf = (_Float16*)ws;  ws += (size_t)BS_ * E_ * sizeof(_Float16); // 8 MB
  _Float16* xh   = (_Float16*)ws;  ws += (size_t)BS_ * E_ * sizeof(_Float16); // 8 MB
  _Float16* Wqh  = (_Float16*)ws;  ws += (size_t)E_ * E_ * sizeof(_Float16);  // 2 MB
  _Float16* Wkh  = (_Float16*)ws;  ws += (size_t)E_ * E_ * sizeof(_Float16);
  _Float16* Wvh  = (_Float16*)ws;  ws += (size_t)E_ * E_ * sizeof(_Float16);
  _Float16* Woh  = (_Float16*)ws;  ws += (size_t)E_ * E_ * sizeof(_Float16);

  // fp32 -> f16 operand conversion (one-time, bandwidth-trivial)
  f32_to_f16_kernel<<<(BS_ * E_) / 1024, 256, 0, stream>>>(x, xh);
  f32_to_f16_kernel<<<(E_ * E_) / 1024, 256, 0, stream>>>(Wq, Wqh);
  f32_to_f16_kernel<<<(E_ * E_) / 1024, 256, 0, stream>>>(Wk, Wkh);
  f32_to_f16_kernel<<<(E_ * E_) / 1024, 256, 0, stream>>>(Wv, Wvh);
  f32_to_f16_kernel<<<(E_ * E_) / 1024, 256, 0, stream>>>(Wo, Woh);

  const dim3 ggrid(E_ / 128, BS_ / 128);  // (8, 32)
  gemm_xwT_kernel<1><<<ggrid, 256, 0, stream>>>(xh, Wqh, bq, (void*)q32, BS_, E_, E_);
  gemm_xwT_kernel<1><<<ggrid, 256, 0, stream>>>(xh, Wkh, bk, (void*)k32, BS_, E_, E_);
  gemm_xwT_kernel<2><<<ggrid, 256, 0, stream>>>(xh, Wvh, bv, (void*)v16p, BS_, E_, E_);

  gates_cumsum_kernel<<<B_ * H_, 256, 0, stream>>>(q32, k32, ak);
  rope_kernel<<<(B_ * H_ * S_ * 32) / 256, 256, 0, stream>>>(q32, k32, ak, qrp, krp);

  flash_attn_kernel<<<dim3(S_ / 128, B_ * H_), 256, 0, stream>>>(qrp, krp, v16p, obuf);

  gemm_xwT_kernel<0><<<ggrid, 256, 0, stream>>>(obuf, Woh, bo, d_out, BS_, E_, E_);
}
